// GaussianUpdater_20229295964872
// MI455X (gfx1250) — compile-verified, matrix operand order checked by reading, not run on hardware
//
#include <hip/hip_runtime.h>

typedef __bf16 bf16;
typedef __attribute__((ext_vector_type(16))) __bf16 bf16x16;
typedef __attribute__((ext_vector_type(8)))  float    f32x8;
typedef __attribute__((ext_vector_type(4)))  unsigned int u32x4;

#define NN  100000     // nodes
#define HID 128
#define K1P 320        // 302 padded to multiple of 32
#define HC  320        // 5 heads * 64 hidden, fused
#define HCP 384        // padded to multiple of 128 for 2-tile-per-wave GEMM
#define NOUT 14        // 3+4+3+3+1 head outputs

// ---------------------------------------------------------------- weights prep
struct PrepArgs {
    const float* g1w; const float* g2w; const float* g3w;
    const float* hw1[5]; const float* hw2[5];
    const float* g1b; const float* g2b; const float* g3b;
    const float* hb1[5]; const float* hb2[5];
    bf16 *w1t, *w2t, *w3t, *wht;     // col-major bf16 (row = output col, len = K)
    float *b123, *bh, *hw2t, *hb2all;
};

__global__ void k_prep(PrepArgs a) {
    const int R0 = 128 * K1P;            // w1t
    const int R1 = 128 * 128;            // w2t
    const int R2 = 128 * 128;            // w3t
    const int R3 = HCP * 128;            // wht (rows 320..383 zero)
    const int R4 = 384;                  // gcn biases
    const int R5 = HCP;                  // head b1 concat (tail zero)
    const int R6 = NOUT * 64;            // head w2 transposed
    const int R7 = NOUT;                 // head b2
    const int TOT = R0 + R1 + R2 + R3 + R4 + R5 + R6 + R7;
    for (int idx = blockIdx.x * blockDim.x + threadIdx.x; idx < TOT;
         idx += gridDim.x * blockDim.x) {
        int t = idx;
        if (t < R0) {                         // w1t[j][k], K padded to 320
            int j = t / K1P, k = t % K1P;
            a.w1t[(size_t)j * K1P + k] = (k < 302) ? (bf16)a.g1w[(size_t)k * 128 + j] : (bf16)0.0f;
            continue;
        } t -= R0;
        if (t < R1) { int j = t / 128, k = t % 128;
            a.w2t[(size_t)j * 128 + k] = (bf16)a.g2w[(size_t)k * 128 + j]; continue; } t -= R1;
        if (t < R2) { int j = t / 128, k = t % 128;
            a.w3t[(size_t)j * 128 + k] = (bf16)a.g3w[(size_t)k * 128 + j]; continue; } t -= R2;
        if (t < R3) {                         // fused head W1: col c -> head c/64, local c%64
            int c = t / 128, k = t % 128;
            bf16 v = (bf16)0.0f;
            if (c < HC) { int hd = c / 64, cc = c % 64;
                          v = (bf16)a.hw1[hd][(size_t)k * 64 + cc]; }
            a.wht[(size_t)c * 128 + k] = v; continue; } t -= R3;
        if (t < R4) {
            a.b123[t] = (t < 128) ? a.g1b[t] : (t < 256) ? a.g2b[t - 128] : a.g3b[t - 256];
            continue; } t -= R4;
        if (t < R5) { a.bh[t] = (t < HC) ? a.hb1[t / 64][t % 64] : 0.0f; continue; } t -= R5;
        if (t < R6) {                         // head W2 -> [o][k], o global out index
            int o = t / 64, k = t % 64;
            int hd, oo, od;
            if      (o < 3)  { hd = 0; oo = o;      od = 3; }
            else if (o < 7)  { hd = 1; oo = o - 3;  od = 4; }
            else if (o < 10) { hd = 2; oo = o - 7;  od = 3; }
            else if (o < 13) { hd = 3; oo = o - 10; od = 3; }
            else             { hd = 4; oo = 0;      od = 1; }
            a.hw2t[o * 64 + k] = a.hw2[hd][(size_t)k * od + oo]; continue; } t -= R6;
        {   int o = t; int hd, oo;
            if      (o < 3)  { hd = 0; oo = o; }
            else if (o < 7)  { hd = 1; oo = o - 3; }
            else if (o < 10) { hd = 2; oo = o - 7; }
            else if (o < 13) { hd = 3; oo = o - 10; }
            else             { hd = 4; oo = 0; }
            a.hb2all[o] = a.hb2[hd][oo];
        }
    }
}

// ---------------------------------------------------------------- feature assembly
__global__ void k_assemble(const float* __restrict__ xyz, const float* __restrict__ rot,
                           const float* __restrict__ color, const float* __restrict__ scale,
                           const float* __restrict__ opacity, const float* __restrict__ x,
                           bf16* __restrict__ featbf) {
    int i = blockIdx.x;
    int c = threadIdx.x;                  // 0..319
    float v;
    if      (c < 288) v = x[(size_t)i * 288 + c];
    else if (c < 291) { int d = c - 288; v = __sinf(xyz[i * 3 + d] * (10.0f * (d + 1))); }
    else if (c < 295) v = rot[i * 4 + (c - 291)];
    else if (c < 298) v = color[i * 3 + (c - 295)];
    else if (c < 301) v = scale[i * 3 + (c - 298)];
    else if (c == 301) v = opacity[i];
    else v = 0.0f;
    featbf[(size_t)i * K1P + c] = (bf16)v;
}

// ---------------------------------------------------------------- CSR build
__global__ void k_zero_i(int* p, int n) {
    int i = blockIdx.x * blockDim.x + threadIdx.x;
    if (i < n) p[i] = 0;
}
__global__ void k_count(const int* __restrict__ col, int* __restrict__ cnt, int e) {
    int i = blockIdx.x * blockDim.x + threadIdx.x;
    if (i < e) atomicAdd(&cnt[col[i]], 1);
}
__global__ void k_dinv(const int* __restrict__ cnt, float* __restrict__ dinv, int n) {
    int i = blockIdx.x * blockDim.x + threadIdx.x;
    if (i < n) dinv[i] = rsqrtf((float)cnt[i] + 1.0f);   // +1 self loop, always > 0
}
// single-block chunked exclusive scan over NN counts -> start[NN+1], cursor copy
__global__ void k_scan(const int* __restrict__ cnt, int* __restrict__ start,
                       int* __restrict__ cursor) {
    __shared__ int s[256];
    __shared__ int carry;
    int tid = threadIdx.x;
    if (tid == 0) carry = 0;
    __syncthreads();
    for (int c0 = 0; c0 < NN; c0 += 256) {
        int i = c0 + tid;
        int v = (i < NN) ? cnt[i] : 0;
        s[tid] = v;
        __syncthreads();
        for (int off = 1; off < 256; off <<= 1) {     // Hillis-Steele inclusive
            int tpl = (tid >= off) ? s[tid - off] : 0;
            __syncthreads();
            s[tid] += tpl;
            __syncthreads();
        }
        if (i < NN) {
            int ex = carry + s[tid] - v;
            start[i] = ex;
            cursor[i] = ex;
        }
        __syncthreads();
        if (tid == 0) carry += s[255];
        __syncthreads();
    }
    if (tid == 0) start[NN] = carry;
}
__global__ void k_bin(const int* __restrict__ row, const int* __restrict__ col,
                      int* __restrict__ cursor, int* __restrict__ csr_src, int e) {
    int i = blockIdx.x * blockDim.x + threadIdx.x;
    if (i < e) {
        int slot = atomicAdd(&cursor[col[i]], 1);
        csr_src[slot] = row[i];
    }
}

// ---------------------------------------------------------------- WMMA GEMM
// C[M,Ncols] = A[M,K](bf16 row-major) x Bt[NcolsPad,K](bf16, B col-major).
// block = 128 thr (4 waves); each wave owns TWO 16x16 col tiles (one A frag,
// two B frags, 2 WMMAs per k-step) -> block = 16 rows x 128 cols.
__global__ void k_gemm_wmma(const bf16* __restrict__ A, const bf16* __restrict__ Bt,
                            int K, int Ncols, const float* __restrict__ bias, int relu,
                            float* __restrict__ outF, bf16* __restrict__ outBF) {
    int wave = threadIdx.x >> 5;
    int lane = threadIdx.x & 31;
    int m    = lane & 15;                 // A row / B col within tile
    int half = lane >> 4;                 // K-half select per 16-bit layout
    int row0 = blockIdx.x * 16;
    int col0 = blockIdx.y * 128 + wave * 32;
    const bf16* Arow  = A  + (size_t)(row0 + m) * K;
    const bf16* Brow0 = Bt + (size_t)(col0 + m) * K;
    const bf16* Brow1 = Bt + (size_t)(col0 + 16 + m) * K;
    union Frag { bf16x16 v; u32x4 q[2]; };
    f32x8 acc0 = {}, acc1 = {};
    for (int k0 = 0; k0 < K; k0 += 32) {
        Frag a, b0, b1;
        a.q[0]  = *reinterpret_cast<const u32x4*>(Arow  + k0 +      half * 8);
        a.q[1]  = *reinterpret_cast<const u32x4*>(Arow  + k0 + 16 + half * 8);
        b0.q[0] = *reinterpret_cast<const u32x4*>(Brow0 + k0 +      half * 8);
        b0.q[1] = *reinterpret_cast<const u32x4*>(Brow0 + k0 + 16 + half * 8);
        b1.q[0] = *reinterpret_cast<const u32x4*>(Brow1 + k0 +      half * 8);
        b1.q[1] = *reinterpret_cast<const u32x4*>(Brow1 + k0 + 16 + half * 8);
        acc0 = __builtin_amdgcn_wmma_f32_16x16x32_bf16(false, a.v, false, b0.v,
                                                       (short)0, acc0, false, false);
        acc1 = __builtin_amdgcn_wmma_f32_16x16x32_bf16(false, a.v, false, b1.v,
                                                       (short)0, acc1, false, false);
    }
#pragma unroll
    for (int tile = 0; tile < 2; ++tile) {
        int ccol = col0 + tile * 16 + m;
        if (ccol >= Ncols) continue;      // padded head columns: skip store only
        float bv = bias ? bias[ccol] : 0.0f;
        const f32x8& acc = tile ? acc1 : acc0;
#pragma unroll
        for (int r = 0; r < 8; ++r) {
            int rr = row0 + r + half * 8; // C layout: vgpr r -> M=r (lo) / M=8+r (hi)
            float v = acc[r] + bv;
            if (relu) v = fmaxf(v, 0.0f);
            if (outF)  outF [(size_t)rr * Ncols + ccol] = v;
            if (outBF) outBF[(size_t)rr * Ncols + ccol] = (bf16)v;
        }
    }
}

// ---------------------------------------------------------------- fused aggregate
// One wave per destination node: register-resident float4 accumulation over
// in-edges (CSR gather, no atomics), + self loop + bias + ReLU -> bf16.
__global__ void k_gather(const float* __restrict__ t, const float* __restrict__ dinv,
                         const int* __restrict__ start, const int* __restrict__ src,
                         const float* __restrict__ bias, bf16* __restrict__ hbf) {
    int node = blockIdx.x * 8 + (threadIdx.x >> 5);
    if (node >= NN) return;
    int lane = threadIdx.x & 31;
    float dc = dinv[node];
    float4 self = reinterpret_cast<const float4*>(t + (size_t)node * HID)[lane];
    float sc = dc * dc;
    float4 acc = make_float4(self.x * sc, self.y * sc, self.z * sc, self.w * sc);
    int s0 = start[node], s1 = start[node + 1];
    for (int k = s0; k < s1; ++k) {
        int r = __builtin_amdgcn_readfirstlane(src[k]);
        if (k + 1 < s1)                               // gfx1250 global_prefetch path
            __builtin_prefetch(t + (size_t)src[k + 1] * HID + lane * 4, 0, 1);
        float nrm = dinv[r] * dc;
        float4 v = reinterpret_cast<const float4*>(t + (size_t)r * HID)[lane];
        acc.x += v.x * nrm; acc.y += v.y * nrm;
        acc.z += v.z * nrm; acc.w += v.w * nrm;
    }
    float4 b = reinterpret_cast<const float4*>(bias)[lane];
    union { bf16 h[4]; uint2 u; } o;
    o.h[0] = (bf16)fmaxf(acc.x + b.x, 0.0f);
    o.h[1] = (bf16)fmaxf(acc.y + b.y, 0.0f);
    o.h[2] = (bf16)fmaxf(acc.z + b.z, 0.0f);
    o.h[3] = (bf16)fmaxf(acc.w + b.w, 0.0f);
    reinterpret_cast<uint2*>(hbf + (size_t)node * HID)[lane] = o.u;
}

// ---------------------------------------------------------------- head second layers
__global__ void k_head2(const bf16* __restrict__ hh, const float* __restrict__ w2t,
                        const float* __restrict__ b2, float* __restrict__ delta) {
    int t = blockIdx.x * blockDim.x + threadIdx.x;
    if (t >= NN * NOUT) return;
    int i = t / NOUT, o = t % NOUT;
    int hd = (o < 3) ? 0 : (o < 7) ? 1 : (o < 10) ? 2 : (o < 13) ? 3 : 4;
    const bf16* h = hh + (size_t)i * HC + hd * 64;
    const float* w = w2t + o * 64;
    float s = b2[o];
#pragma unroll 8
    for (int k = 0; k < 64; ++k) s += (float)h[k] * w[k];
    delta[(size_t)i * NOUT + o] = s;
}

// ---------------------------------------------------------------- postprocess
__global__ void k_post(const float* __restrict__ xyz, const float* __restrict__ rot,
                       const float* __restrict__ color, const float* __restrict__ scale,
                       const float* __restrict__ opacity, const float* __restrict__ delta,
                       float* __restrict__ out) {
    int i = blockIdx.x * blockDim.x + threadIdx.x;
    if (i >= NN) return;
    const float* d = delta + (size_t)i * NOUT;
    float* o_xyz = out;
    float* o_rot = out + (size_t)3 * NN;
    float* o_col = out + (size_t)7 * NN;
    float* o_scl = out + (size_t)10 * NN;
    float* o_opa = out + (size_t)13 * NN;
    for (int j = 0; j < 3; ++j)
        o_xyz[i * 3 + j] = xyz[i * 3 + j] + d[j] * 0.1f;
    float r[4]; float nrm = 0.0f;
    for (int j = 0; j < 4; ++j) { r[j] = rot[i * 4 + j] + d[3 + j]; nrm += r[j] * r[j]; }
    nrm = fmaxf(sqrtf(nrm), 1e-12f);
    for (int j = 0; j < 4; ++j) o_rot[i * 4 + j] = r[j] / nrm;
    for (int j = 0; j < 3; ++j)
        o_col[i * 3 + j] = fminf(fmaxf(color[i * 3 + j] + d[7 + j], 0.0f), 1.0f);
    for (int j = 0; j < 3; ++j)
        o_scl[i * 3 + j] = fminf(fmaxf(scale[i * 3 + j] + d[10 + j] * 0.1f, 1e-6f), 1000.0f);
    o_opa[i] = fminf(fmaxf(opacity[i] + d[13], 0.0f), 1.0f);
}

// ---------------------------------------------------------------- host driver
extern "C" void kernel_launch(void* const* d_in, const int* in_sizes, int n_in,
                              void* d_out, int out_size, void* d_ws, size_t ws_size,
                              hipStream_t stream) {
    const float* xyz     = (const float*)d_in[0];
    const float* rot     = (const float*)d_in[1];
    const float* color   = (const float*)d_in[2];
    const float* scale   = (const float*)d_in[3];
    const float* opacity = (const float*)d_in[4];
    const float* x       = (const float*)d_in[5];
    const int*   ei      = (const int*)d_in[6];
    const int E = in_sizes[6] / 2;
    const int* erow = ei;
    const int* ecol = ei + E;

    // params: insertion order gcn1,gcn2,gcn3, heads pos,rotd,col,scl,opa; each (w,b)/((w1,b1),(w2,b2))
    int p = 7;
    const float* g1w = (const float*)d_in[p++]; const float* g1b = (const float*)d_in[p++];
    const float* g2w = (const float*)d_in[p++]; const float* g2b = (const float*)d_in[p++];
    const float* g3w = (const float*)d_in[p++]; const float* g3b = (const float*)d_in[p++];
    const float *hw1[5], *hb1[5], *hw2[5], *hb2[5];
    for (int h = 0; h < 5; ++h) {
        hw1[h] = (const float*)d_in[p++]; hb1[h] = (const float*)d_in[p++];
        hw2[h] = (const float*)d_in[p++]; hb2[h] = (const float*)d_in[p++];
    }

    // workspace carve-out (256B aligned)
    char* wp = (char*)d_ws;
    auto alloc = [&](size_t bytes) { void* q = wp; wp += (bytes + 255) & ~(size_t)255; return q; };
    bf16*  featbf = (bf16*) alloc((size_t)NN * K1P * 2);
    bf16*  hbf    = (bf16*) alloc((size_t)NN * HID * 2);
    bf16*  hhbf   = (bf16*) alloc((size_t)NN * HC  * 2);
    float* t      = (float*)alloc((size_t)NN * HID * 4);
    float* delta  = (float*)alloc((size_t)NN * NOUT * 4);
    int*   cnt    = (int*)  alloc((size_t)NN * 4);
    float* dinv   = (float*)alloc((size_t)NN * 4);
    int*   start  = (int*)  alloc((size_t)(NN + 1) * 4);
    int*   cursor = (int*)  alloc((size_t)NN * 4);
    int*   csrsrc = (int*)  alloc((size_t)E * 4);
    bf16*  w1t    = (bf16*) alloc((size_t)128 * K1P * 2);
    bf16*  w2t    = (bf16*) alloc((size_t)128 * 128 * 2);
    bf16*  w3t    = (bf16*) alloc((size_t)128 * 128 * 2);
    bf16*  wht    = (bf16*) alloc((size_t)HCP * 128 * 2);
    float* b123   = (float*)alloc(384 * 4);
    float* bh     = (float*)alloc(HCP * 4);
    float* hw2t   = (float*)alloc(NOUT * 64 * 4);
    float* hb2all = (float*)alloc(NOUT * 4);

    PrepArgs pa;
    pa.g1w = g1w; pa.g2w = g2w; pa.g3w = g3w;
    pa.g1b = g1b; pa.g2b = g2b; pa.g3b = g3b;
    for (int h = 0; h < 5; ++h) { pa.hw1[h] = hw1[h]; pa.hw2[h] = hw2[h];
                                  pa.hb1[h] = hb1[h]; pa.hb2[h] = hb2[h]; }
    pa.w1t = w1t; pa.w2t = w2t; pa.w3t = w3t; pa.wht = wht;
    pa.b123 = b123; pa.bh = bh; pa.hw2t = hw2t; pa.hb2all = hb2all;
    k_prep<<<512, 256, 0, stream>>>(pa);

    k_assemble<<<NN, K1P, 0, stream>>>(xyz, rot, color, scale, opacity, x, featbf);

    // CSR build: count -> dinv -> scan -> bin
    k_zero_i<<<(NN + 255) / 256, 256, 0, stream>>>(cnt, NN);
    k_count<<<(E + 255) / 256, 256, 0, stream>>>(ecol, cnt, E);
    k_dinv<<<(NN + 255) / 256, 256, 0, stream>>>(cnt, dinv, NN);
    k_scan<<<1, 256, 0, stream>>>(cnt, start, cursor);
    k_bin<<<(E + 255) / 256, 256, 0, stream>>>(erow, ecol, cursor, csrsrc, E);

    dim3 gemm_grid(NN / 16, 1);          // 128 cols per block
    dim3 gather_grid((NN + 7) / 8);
    // layer 1
    k_gemm_wmma<<<gemm_grid, 128, 0, stream>>>(featbf, w1t, K1P, HID, nullptr, 0, t, nullptr);
    k_gather<<<gather_grid, 256, 0, stream>>>(t, dinv, start, csrsrc, b123, hbf);
    // layer 2
    k_gemm_wmma<<<gemm_grid, 128, 0, stream>>>(hbf, w2t, HID, HID, nullptr, 0, t, nullptr);
    k_gather<<<gather_grid, 256, 0, stream>>>(t, dinv, start, csrsrc, b123 + 128, hbf);
    // layer 3
    k_gemm_wmma<<<gemm_grid, 128, 0, stream>>>(hbf, w3t, HID, HID, nullptr, 0, t, nullptr);
    k_gather<<<gather_grid, 256, 0, stream>>>(t, dinv, start, csrsrc, b123 + 256, hbf);

    // fused head first layers: [N,128] x [128,320(pad 384)] + bias, ReLU, bf16 out
    dim3 head_grid(NN / 16, HCP / 128);  // 6250 x 3
    k_gemm_wmma<<<head_grid, 128, 0, stream>>>(hbf, wht, HID, HC, bh, 1, nullptr, hhbf);
    k_head2<<<(NN * NOUT + 255) / 256, 256, 0, stream>>>(hhbf, hw2t, hb2all, delta);
    k_post<<<(NN + 255) / 256, 256, 0, stream>>>(xyz, rot, color, scale, opacity, delta,
                                                 (float*)d_out);
}
